// CNF_ODE_78108275245410
// MI455X (gfx1250) — matrix operand order repaired and checked by reading, not compiled, not run on hardware
//
#include <hip/hip_runtime.h>
#include <cstddef>

#define NSAMP   1048576
#define TILES   (NSAMP / 16)
#define WAVES   8
#define THREADS (WAVES * 32)

typedef __attribute__((ext_vector_type(16))) _Float16 v16h;
typedef __attribute__((ext_vector_type(8)))  _Float16 v8h;
typedef __attribute__((ext_vector_type(8)))  float    v8f;

__device__ __forceinline__ v16h cat16(v8h lo, v8h hi) {
  return __builtin_shufflevector(lo, hi, 0,1,2,3,4,5,6,7,8,9,10,11,12,13,14,15);
}

// A fragment: 16(M) x 32(K) f16 from row-major LDS tile [16][stride].
// ISA layout: lane m=l&15; halves v0..v3 = K(hi*8..hi*8+7), v4..v7 = K(16+hi*8..+7)
__device__ __forceinline__ v16h ld_a(const _Float16* t, int stride, int ko, int lane) {
  const int m = lane & 15, hi = lane >> 4;
  const _Float16* p = t + m * stride + ko + hi * 8;
  v8h lo = *(const v8h*)(p);
  v8h hv = *(const v8h*)(p + 16);
  return cat16(lo, hv);
}

// B fragment: 32(K) x 16(N) f16 from LDS tile stored as [N][stride] (i.e. B^T row-major).
// ISA layout: lane n=l&15; halves v0..v3 = K(hi*8..hi*8+7), v4..v7 = K(16+hi*8..+7)
__device__ __forceinline__ v16h ld_b(const _Float16* t, int stride, int ko, int lane) {
  const int n = lane & 15, hi = lane >> 4;
  const _Float16* p = t + n * stride + ko + hi * 8;
  v8h lo = *(const v8h*)(p);
  v8h hv = *(const v8h*)(p + 16);
  return cat16(lo, hv);
}

__device__ __forceinline__ v8f wmma32(v16h a, v16h b, v8f c) {
  return __builtin_amdgcn_wmma_f32_16x16x32_f16(false, a, false, b, (short)0, c, false, false);
}

// Fast sigmoid: v_exp_f32 + v_rcp_f32 (no IEEE divide expansion).
__device__ __forceinline__ float fast_sigmoid(float z) {
  return __builtin_amdgcn_rcpf(1.0f + __expf(-z));
}

// D layout: lane col n=l&15, VGPR r <-> row (r + 8*hi).  Apply bias+SiLU, store h and silu'.
__device__ __forceinline__ void act_store(v8f acc, const float* bias, _Float16* H, _Float16* Dd,
                                          int stride, int col, int hiL) {
  const float bb = bias[col];
#pragma unroll
  for (int r = 0; r < 8; ++r) {
    const float z = acc[r] + bb;
    const float s = fast_sigmoid(z);
    const float h = z * s;
    const float d = s * (1.0f + z - h);   // s*(1 + z*(1-s))
    const int mm = r + 8 * hiL;
    H [mm * stride + col] = (_Float16)h;
    Dd[mm * stride + col] = (_Float16)d;
  }
}

// gz = g * silu'(z), stored back as f16 for the next backward matmul.
__device__ __forceinline__ void grad_store(v8f acc, const _Float16* Dd, _Float16* GZ,
                                           int stride, int col, int hiL) {
#pragma unroll
  for (int r = 0; r < 8; ++r) {
    const int mm = r + 8 * hiL;
    const float g = acc[r] * (float)Dd[mm * stride + col];
    GZ[mm * stride + col] = (_Float16)g;
  }
}

__global__ __launch_bounds__(THREADS)
void CNF_ODE_78108275245410_kernel(const float* __restrict__ tptr, const float* __restrict__ y,
                                   const float* __restrict__ v,
                                   const float* __restrict__ W1, const float* __restrict__ b1,
                                   const float* __restrict__ W2, const float* __restrict__ b2,
                                   const float* __restrict__ W3, const float* __restrict__ b3,
                                   const float* __restrict__ W4, const float* __restrict__ b4,
                                   float* __restrict__ out) {
  // ---- forward weight tiles (B^T row-major [N][K], f16) ----
  __shared__ __align__(16) _Float16 sW1T[128 * 32];    // W1 [4,128] -> [n=128][k=32], k>=4 zero
  __shared__ __align__(16) _Float16 sW2T[128 * 128];   // W2 [128,128] transposed
  __shared__ __align__(16) _Float16 sW3T[64 * 128];    // W3 [128,64] transposed
  __shared__ __align__(16) _Float16 sW4T[16 * 64];     // W4 [64,2] -> [n=16][k=64], n>=2 zero
  // ---- backward weight tiles ((W^T)^T = W row-major) ----
  __shared__ __align__(16) _Float16 sBW4[64 * 32];     // W4 rows, k>=2 zero
  __shared__ __align__(16) _Float16 sBW3[128 * 64];    // W3 rows
  __shared__ __align__(16) _Float16 sBW2[128 * 128];   // W2 rows
  __shared__ __align__(16) _Float16 sBW1[16 * 128];    // W1 rows, n>=4 zero
  __shared__ float sB1[128], sB2[128], sB3[64], sB4[2];
  // ---- per-wave activation tiles (f16, row-major [16][stride]) ----
  __shared__ __align__(16) _Float16 sX [WAVES][16 * 32];
  __shared__ __align__(16) _Float16 sU [WAVES][16 * 32];
  __shared__ __align__(16) _Float16 sH1[WAVES][16 * 128];  // reused as GZ1
  __shared__ __align__(16) _Float16 sD1[WAVES][16 * 128];
  __shared__ __align__(16) _Float16 sH2[WAVES][16 * 128];  // reused as GZ2
  __shared__ __align__(16) _Float16 sD2[WAVES][16 * 128];
  __shared__ __align__(16) _Float16 sH3[WAVES][16 * 64];   // reused as GZ3
  __shared__ __align__(16) _Float16 sD3[WAVES][16 * 64];
  __shared__ float sSG[WAVES][16 * 4];

  const int tid = threadIdx.x;

  // ---------- cooperative weight staging (once per workgroup) ----------
  for (int i = tid; i < 128 * 32; i += THREADS) { int n = i >> 5, k = i & 31;
    sW1T[i] = (_Float16)(k < 4 ? W1[k * 128 + n] : 0.0f); }
  for (int i = tid; i < 128 * 128; i += THREADS) { int n = i >> 7, k = i & 127;
    sW2T[i] = (_Float16)W2[k * 128 + n]; }
  for (int i = tid; i < 64 * 128; i += THREADS) { int n = i >> 7, k = i & 127;
    sW3T[i] = (_Float16)W3[k * 64 + n]; }
  for (int i = tid; i < 16 * 64; i += THREADS) { int n = i >> 6, k = i & 63;
    sW4T[i] = (_Float16)(n < 2 ? W4[k * 2 + n] : 0.0f); }
  for (int i = tid; i < 64 * 32; i += THREADS) { int n = i >> 5, k = i & 31;
    sBW4[i] = (_Float16)(k < 2 ? W4[n * 2 + k] : 0.0f); }
  for (int i = tid; i < 128 * 64; i += THREADS)  sBW3[i] = (_Float16)W3[i];
  for (int i = tid; i < 128 * 128; i += THREADS) sBW2[i] = (_Float16)W2[i];
  for (int i = tid; i < 16 * 128; i += THREADS) { int n = i >> 7, k = i & 127;
    sBW1[i] = (_Float16)(n < 4 ? W1[n * 128 + k] : 0.0f); }
  for (int i = tid; i < 128; i += THREADS) { sB1[i] = b1[i]; sB2[i] = b2[i]; }
  for (int i = tid; i < 64;  i += THREADS) sB3[i] = b3[i];
  if (tid < 2) sB4[tid] = b4[tid];
  __syncthreads();

  const float tval = tptr[0];
  const int lane = tid & 31;
  const int w    = tid >> 5;
  const int m    = lane & 15;     // sample row owned for A-frags / staging
  const int hiL  = lane >> 4;
  const int col  = lane & 15;     // D-layout column for this lane

  _Float16* X  = sX[w];  _Float16* U  = sU[w];
  _Float16* H1 = sH1[w]; _Float16* D1 = sD1[w];
  _Float16* H2 = sH2[w]; _Float16* D2 = sD2[w];
  _Float16* H3 = sH3[w]; _Float16* D3 = sD3[w];

  const int waveStride = gridDim.x * WAVES;
  for (int tile = blockIdx.x * WAVES + w; tile < TILES; tile += waveStride) {
    const int s0 = tile * 16;
    if (tile + waveStride < TILES) {
      __builtin_prefetch(y + (size_t)(tile + waveStride) * 48, 0, 0);
      __builtin_prefetch(v + (size_t)(tile + waveStride) * 48, 0, 0);
    }

    // ---------- stage inputs: X = [y0 y1 y2 t 0...], U = [v0 v1 0...] ----------
    {
      _Float16* xr = X + m * 32 + hiL * 16;
      _Float16* ur = U + m * 32 + hiL * 16;
      v8h zz = {};
      *(v8h*)xr = zz; *(v8h*)(xr + 8) = zz;
      *(v8h*)ur = zz; *(v8h*)(ur + 8) = zz;
      if (hiL == 0) {
        const float* yr = y + (size_t)(s0 + m) * 3;
        const float* vr = v + (size_t)(s0 + m) * 3;
        xr[0] = (_Float16)yr[0]; xr[1] = (_Float16)yr[1];
        xr[2] = (_Float16)yr[2]; xr[3] = (_Float16)tval;
        ur[0] = (_Float16)vr[0]; ur[1] = (_Float16)vr[1];
      }
    }

    // ---------- layer 1: Z1[16,128] = X[16,4] @ W1 ----------
    {
      v16h a = ld_a(X, 32, 0, lane);
#pragma unroll
      for (int nt = 0; nt < 8; ++nt) {
        v8f acc = {};
        acc = wmma32(a, ld_b(sW1T + nt * 16 * 32, 32, 0, lane), acc);
        act_store(acc, sB1 + nt * 16, H1 + nt * 16, D1 + nt * 16, 128, col, hiL);
      }
    }
    // ---------- layer 2: Z2[16,128] = H1 @ W2 ----------
    {
      v16h a0 = ld_a(H1, 128, 0, lane),  a1 = ld_a(H1, 128, 32, lane);
      v16h a2 = ld_a(H1, 128, 64, lane), a3 = ld_a(H1, 128, 96, lane);
#pragma unroll
      for (int nt = 0; nt < 8; ++nt) {
        const _Float16* bt = sW2T + nt * 16 * 128;
        v8f acc = {};
        acc = wmma32(a0, ld_b(bt, 128, 0, lane), acc);
        acc = wmma32(a1, ld_b(bt, 128, 32, lane), acc);
        acc = wmma32(a2, ld_b(bt, 128, 64, lane), acc);
        acc = wmma32(a3, ld_b(bt, 128, 96, lane), acc);
        act_store(acc, sB2 + nt * 16, H2 + nt * 16, D2 + nt * 16, 128, col, hiL);
      }
    }
    // ---------- layer 3: Z3[16,64] = H2 @ W3 ----------
    {
      v16h a0 = ld_a(H2, 128, 0, lane),  a1 = ld_a(H2, 128, 32, lane);
      v16h a2 = ld_a(H2, 128, 64, lane), a3 = ld_a(H2, 128, 96, lane);
#pragma unroll
      for (int nt = 0; nt < 4; ++nt) {
        const _Float16* bt = sW3T + nt * 16 * 128;
        v8f acc = {};
        acc = wmma32(a0, ld_b(bt, 128, 0, lane), acc);
        acc = wmma32(a1, ld_b(bt, 128, 32, lane), acc);
        acc = wmma32(a2, ld_b(bt, 128, 64, lane), acc);
        acc = wmma32(a3, ld_b(bt, 128, 96, lane), acc);
        act_store(acc, sB3 + nt * 16, H3 + nt * 16, D3 + nt * 16, 64, col, hiL);
      }
    }
    // ---------- layer 4: OUT[16,2] = H3 @ W4  (N padded to 16; cols>=2 are zero) ----------
    {
      v16h a0 = ld_a(H3, 64, 0, lane), a1 = ld_a(H3, 64, 32, lane);
      v8f acc = {};
      acc = wmma32(a0, ld_b(sW4T, 64, 0, lane), acc);
      acc = wmma32(a1, ld_b(sW4T, 64, 32, lane), acc);
      if (col < 3) {
        const float bb = (col < 2) ? sB4[col] : 0.0f;
#pragma unroll
        for (int r = 0; r < 8; ++r) {
          const int mm = r + 8 * hiL;
          out[(size_t)(s0 + mm) * 3 + col] = (col < 2) ? (acc[r] + bb) : 0.0f;
        }
      }
    }

    // ---------- backward: G3[16,64] = U[16,2] @ W4^T ; GZ3 = G3 * silu'(Z3) ----------
    {
      v16h a = ld_a(U, 32, 0, lane);
#pragma unroll
      for (int nt = 0; nt < 4; ++nt) {
        v8f acc = {};
        acc = wmma32(a, ld_b(sBW4 + nt * 16 * 32, 32, 0, lane), acc);
        grad_store(acc, D3 + nt * 16, H3 + nt * 16, 64, col, hiL);  // H3 now holds GZ3
      }
    }
    // ---------- G2[16,128] = GZ3 @ W3^T ; GZ2 = G2 * silu'(Z2) ----------
    {
      v16h a0 = ld_a(H3, 64, 0, lane), a1 = ld_a(H3, 64, 32, lane);
#pragma unroll
      for (int nt = 0; nt < 8; ++nt) {
        const _Float16* bt = sBW3 + nt * 16 * 64;
        v8f acc = {};
        acc = wmma32(a0, ld_b(bt, 64, 0, lane), acc);
        acc = wmma32(a1, ld_b(bt, 64, 32, lane), acc);
        grad_store(acc, D2 + nt * 16, H2 + nt * 16, 128, col, hiL); // H2 now holds GZ2
      }
    }
    // ---------- G1[16,128] = GZ2 @ W2^T ; GZ1 = G1 * silu'(Z1) ----------
    {
      v16h a0 = ld_a(H2, 128, 0, lane),  a1 = ld_a(H2, 128, 32, lane);
      v16h a2 = ld_a(H2, 128, 64, lane), a3 = ld_a(H2, 128, 96, lane);
#pragma unroll
      for (int nt = 0; nt < 8; ++nt) {
        const _Float16* bt = sBW2 + nt * 16 * 128;
        v8f acc = {};
        acc = wmma32(a0, ld_b(bt, 128, 0, lane), acc);
        acc = wmma32(a1, ld_b(bt, 128, 32, lane), acc);
        acc = wmma32(a2, ld_b(bt, 128, 64, lane), acc);
        acc = wmma32(a3, ld_b(bt, 128, 96, lane), acc);
        grad_store(acc, D1 + nt * 16, H1 + nt * 16, 128, col, hiL); // H1 now holds GZ1
      }
    }
    // ---------- GX[16,4] = GZ1 @ W1^T (N padded to 16); div = sum_n<3 GX*v ----------
    {
      v16h a0 = ld_a(H1, 128, 0, lane),  a1 = ld_a(H1, 128, 32, lane);
      v16h a2 = ld_a(H1, 128, 64, lane), a3 = ld_a(H1, 128, 96, lane);
      v8f acc = {};
      acc = wmma32(a0, ld_b(sBW1, 128, 0, lane), acc);
      acc = wmma32(a1, ld_b(sBW1, 128, 32, lane), acc);
      acc = wmma32(a2, ld_b(sBW1, 128, 64, lane), acc);
      acc = wmma32(a3, ld_b(sBW1, 128, 96, lane), acc);
      if (col < 4) {
#pragma unroll
        for (int r = 0; r < 8; ++r) sSG[w][(r + 8 * hiL) * 4 + col] = acc[r];
      }
      if (lane < 16) {  // same-wave LDS is processed in order
        const float* vr = v + (size_t)(s0 + lane) * 3;
        const float dv = sSG[w][lane * 4 + 0] * vr[0]
                       + sSG[w][lane * 4 + 1] * vr[1]
                       + sSG[w][lane * 4 + 2] * vr[2];
        out[(size_t)3 * NSAMP + s0 + lane] = -dv;
      }
    }
  }
}

extern "C" void kernel_launch(void* const* d_in, const int* in_sizes, int n_in,
                              void* d_out, int out_size, void* d_ws, size_t ws_size,
                              hipStream_t stream) {
  (void)in_sizes; (void)n_in; (void)out_size; (void)d_ws; (void)ws_size;
  const float* t  = (const float*)d_in[0];
  const float* y  = (const float*)d_in[1];
  // d_in[2] = logp (unused by the reference computation)
  const float* v  = (const float*)d_in[3];
  const float* W1 = (const float*)d_in[4];
  const float* b1 = (const float*)d_in[5];
  const float* W2 = (const float*)d_in[6];
  const float* b2 = (const float*)d_in[7];
  const float* W3 = (const float*)d_in[8];
  const float* b3 = (const float*)d_in[9];
  const float* W4 = (const float*)d_in[10];
  const float* b4 = (const float*)d_in[11];
  CNF_ODE_78108275245410_kernel<<<dim3(1024), dim3(THREADS), 0, stream>>>(
      t, y, v, W1, b1, W2, b2, W3, b3, W4, b4, (float*)d_out);
}
